// HyperGNN_35519379537926
// MI455X (gfx1250) — matrix operand order repaired and checked by reading, not compiled, not run on hardware
//
#include <hip/hip_runtime.h>
#include <hip/hip_fp16.h>

#define N_NODES  100000
#define NNZ      1600000
#define N_EDGES  100000
#define HID      64
#define OUT_D    16
#define N_GRAPHS 64

typedef __attribute__((ext_vector_type(2)))  float    v2f;
typedef __attribute__((ext_vector_type(8)))  float    v8f;
typedef __attribute__((ext_vector_type(16))) _Float16 v16h;

// ---------------------------------------------------------------------------
// Y[r, 0:64] = X[r, 0:64] @ W[64, 64]   (row-major), via WMMA.
// 256 threads = 8 waves; each wave computes a 16x64 output strip.
// ---------------------------------------------------------------------------
__global__ __launch_bounds__(256) void gemm64_wmma(const float* __restrict__ X,
                                                   const float* __restrict__ W,
                                                   float* __restrict__ Y,
                                                   int nrows) {
    __shared__ float sW[64 * 64];
    const int tid = threadIdx.x;
    for (int i = tid; i < 64 * 64 / 4; i += 256)
        ((float4*)sW)[i] = ((const float4*)W)[i];
    __syncthreads();

    const int wave    = tid >> 5;
    const int lane    = tid & 31;
    const int rowBase = blockIdx.x * 128 + wave * 16;
    if (rowBase >= nrows) return;          // wave-uniform guard (EXEC stays all-1s)

    const int m  = lane & 15;              // output column within 16-tile / A row
    const int hi = lane >> 4;              // lane half

    const float* xrow = X + (size_t)(rowBase + m) * 64;
    v8f c[4] = {v8f{}, v8f{}, v8f{}, v8f{}};

#if __has_builtin(__builtin_amdgcn_wmma_f32_16x16x4_f32)
    // f32 path: A 16x4 = 2 VGPRs/lane; lane m (half hi): K = k0 + 2*hi + {0,1}
    for (int k0 = 0; k0 < 64; k0 += 4) {
        v2f a;
        a[0] = xrow[k0 + 2 * hi + 0];
        a[1] = xrow[k0 + 2 * hi + 1];
#pragma unroll
        for (int t = 0; t < 4; ++t) {
            v2f b;
            b[0] = sW[(k0 + 2 * hi + 0) * 64 + t * 16 + m];
            b[1] = sW[(k0 + 2 * hi + 1) * 64 + t * 16 + m];
            c[t] = __builtin_amdgcn_wmma_f32_16x16x4_f32(
                false, a, false, b, (short)0, c[t], false, false);
        }
    }
#else
    // Fallback: f16 inputs, f32 accumulation (codegen-confirmed builtin).
    for (int kb = 0; kb < 64; kb += 32) {
        v16h a;
#pragma unroll
        for (int j = 0; j < 8; ++j) {
            a[j]     = (_Float16)xrow[kb + 16 * hi + j];
            a[j + 8] = (_Float16)xrow[kb + 16 * hi + 8 + j];
        }
#pragma unroll
        for (int t = 0; t < 4; ++t) {
            v16h b;
#pragma unroll
            for (int j = 0; j < 16; ++j)
                b[j] = (_Float16)sW[(kb + 16 * hi + j) * 64 + t * 16 + m];
            c[t] = __builtin_amdgcn_wmma_f32_16x16x32_f16(
                false, a, false, b, (short)0, c[t], false, false);
        }
    }
#endif

    // C/D layout: VGPR v, lane half hi -> row (v + 8*hi), col = t*16 + m
#pragma unroll
    for (int t = 0; t < 4; ++t)
#pragma unroll
        for (int v = 0; v < 8; ++v)
            Y[(size_t)(rowBase + v + 8 * hi) * 64 + t * 16 + m] = c[t][v];
}

// ---------------------------------------------------------------------------
// Degrees: D[node]+=1, Bdeg[edge]+=1 over all incidences.
// ---------------------------------------------------------------------------
__global__ void deg_kernel(const int* __restrict__ ni, const int* __restrict__ ei,
                           float* __restrict__ D, float* __restrict__ B) {
    int i = blockIdx.x * blockDim.x + threadIdx.x;
    if (i >= NNZ) return;
    atomicAdd(&D[ni[i]], 1.0f);
    atomicAdd(&B[ei[i]], 1.0f);
}

__global__ void invert_kernel(float* __restrict__ D, float* __restrict__ B) {
    int i = blockIdx.x * blockDim.x + threadIdx.x;
    if (i < N_NODES) { float d = D[i]; D[i] = d > 0.0f ? 1.0f / d : 0.0f; }
    if (i < N_EDGES) { float b = B[i]; B[i] = b > 0.0f ? 1.0f / b : 0.0f; }
}

// ---------------------------------------------------------------------------
// node -> hyperedge scatter: dst[edge,:] += src[node,:]
// 16 threads per incidence, float4 each (half-wave = one 256B row).
// ---------------------------------------------------------------------------
__global__ __launch_bounds__(256) void edge_scatter(const float* __restrict__ src,
                                                    const int* __restrict__ ni,
                                                    const int* __restrict__ ei,
                                                    float* __restrict__ dst) {
    long long gid = (long long)blockIdx.x * blockDim.x + threadIdx.x;
    int i = (int)(gid >> 4);
    if (i >= NNZ) return;
    int c = ((int)gid & 15) * 4;
    int n = ni[i], e = ei[i];
    float4 v = *(const float4*)(src + (size_t)n * 64 + c);
    float* d = dst + (size_t)e * 64 + c;
    atomicAdd(d + 0, v.x); atomicAdd(d + 1, v.y);
    atomicAdd(d + 2, v.z); atomicAdd(d + 3, v.w);
}

// ---------------------------------------------------------------------------
// hyperedge -> node scatter with Binv folded in: dst[node,:] += Binv[e]*ef[e,:]
// ---------------------------------------------------------------------------
__global__ __launch_bounds__(256) void node_scatter(const float* __restrict__ ef,
                                                    const float* __restrict__ Binv,
                                                    const int* __restrict__ ni,
                                                    const int* __restrict__ ei,
                                                    float* __restrict__ dst) {
    long long gid = (long long)blockIdx.x * blockDim.x + threadIdx.x;
    int i = (int)(gid >> 4);
    if (i >= NNZ) return;
    int c = ((int)gid & 15) * 4;
    int n = ni[i], e = ei[i];
    float s = Binv[e];
    float4 v = *(const float4*)(ef + (size_t)e * 64 + c);
    float* d = dst + (size_t)n * 64 + c;
    atomicAdd(d + 0, s * v.x); atomicAdd(d + 1, s * v.y);
    atomicAdd(d + 2, s * v.z); atomicAdd(d + 3, s * v.w);
}

// h1 = relu(Dinv * agg1 + b1), in place.
__global__ void finalize1(float* __restrict__ A, const float* __restrict__ Dinv,
                          const float* __restrict__ b1) {
    int idx = blockIdx.x * blockDim.x + threadIdx.x;
    if (idx >= N_NODES * 64) return;
    int i = idx >> 6, d = idx & 63;
    float v = Dinv[i] * A[idx] + b1[d];
    A[idx] = v > 0.0f ? v : 0.0f;
}

// h = relu(h1 + (Dinv*agg2 + b2)); per-graph sum (f32 atomics) + max (int-reinterp).
__global__ void finalize2_pool(const float* __restrict__ h1, const float* __restrict__ agg2,
                               const float* __restrict__ Dinv, const float* __restrict__ b2,
                               const int* __restrict__ batch,
                               float* __restrict__ gsum, int* __restrict__ gmax) {
    int idx = blockIdx.x * blockDim.x + threadIdx.x;
    if (idx >= N_NODES * 64) return;
    int i = idx >> 6, d = idx & 63;
    float h2 = Dinv[i] * agg2[idx] + b2[d];
    float h = h1[idx] + h2;
    h = h > 0.0f ? h : 0.0f;
    int g = batch[i];
    atomicAdd(&gsum[g * 64 + d], h);
    atomicMax(&gmax[g * 64 + d], __float_as_int(h));   // valid: h >= 0, init 0
}

__global__ void count_kernel(const int* __restrict__ batch, float* __restrict__ cnt) {
    int i = blockIdx.x * blockDim.x + threadIdx.x;
    if (i >= N_NODES) return;
    atomicAdd(&cnt[batch[i]], 1.0f);
}

// out[g, 0:16] = [mean | max | sum] @ Wlin + blin. One block of 1024 threads.
__global__ __launch_bounds__(1024) void head_kernel(const float* __restrict__ gsum,
                                                    const int* __restrict__ gmax,
                                                    const float* __restrict__ cnt,
                                                    const float* __restrict__ Wlin,
                                                    const float* __restrict__ blin,
                                                    float* __restrict__ out) {
    int tid = threadIdx.x;
    int g = tid >> 4, o = tid & 15;
    float c = cnt[g]; c = c > 1.0f ? c : 1.0f;
    float acc = blin[o];
    for (int k = 0; k < 64; ++k) {
        float sv = gsum[g * 64 + k];
        float mv = __int_as_float(gmax[g * 64 + k]);
        acc += (sv / c) * Wlin[k * 16 + o]
             + mv       * Wlin[(64 + k) * 16 + o]
             + sv       * Wlin[(128 + k) * 16 + o];
    }
    out[g * 16 + o] = acc;
}

extern "C" void kernel_launch(void* const* d_in, const int* in_sizes, int n_in,
                              void* d_out, int out_size, void* d_ws, size_t ws_size,
                              hipStream_t stream) {
    const float* x     = (const float*)d_in[0];
    const int*   hei   = (const int*)d_in[1];
    const int*   ni    = hei;            // node_idx = hyperedge_index[0]
    const int*   ei    = hei + NNZ;      // edge_idx = hyperedge_index[1]
    const int*   batch = (const int*)d_in[2];
    const float* W1    = (const float*)d_in[3];
    const float* b1    = (const float*)d_in[4];
    const float* W2    = (const float*)d_in[5];
    const float* b2    = (const float*)d_in[6];
    const float* Wlin  = (const float*)d_in[7];
    const float* blin  = (const float*)d_in[8];
    float*       out   = (float*)d_out;

    const size_t FEAT = (size_t)N_NODES * 64;   // 6.4M floats
    const size_t EF   = (size_t)N_EDGES * 64;   // 6.4M floats

    float* A    = (float*)d_ws;          // xw1 -> agg1 -> h1
    float* B    = A + FEAT;              // xw2 -> agg2
    float* E    = B + FEAT;              // e_feat (both convs)
    float* Dinv = E + EF;
    float* Binv = Dinv + N_NODES;
    float* cnt  = Binv + N_EDGES;
    float* gsum = cnt + N_GRAPHS;
    int*   gmax = (int*)(gsum + (size_t)N_GRAPHS * 64);

    // Zero Dinv..gmax (contiguous) in one memset.
    size_t smallBytes = (size_t)(N_NODES + N_EDGES + N_GRAPHS + N_GRAPHS * 64 + N_GRAPHS * 64) * 4;
    hipMemsetAsync(Dinv, 0, smallBytes, stream);

    // Degrees and inverses.
    deg_kernel<<<(NNZ + 255) / 256, 256, 0, stream>>>(ni, ei, Dinv, Binv);
    invert_kernel<<<(N_NODES + 255) / 256, 256, 0, stream>>>(Dinv, Binv);
    count_kernel<<<(N_NODES + 255) / 256, 256, 0, stream>>>(batch, cnt);

    const int gemmGrid    = (N_NODES + 127) / 128;
    const int scatterGrid = (int)(((long long)NNZ * 16 + 255) / 256);
    const int featGrid    = (int)((FEAT + 255) / 256);

    // ---- conv1 ----
    gemm64_wmma<<<gemmGrid, 256, 0, stream>>>(x, W1, A, N_NODES);
    hipMemsetAsync(E, 0, EF * 4, stream);
    edge_scatter<<<scatterGrid, 256, 0, stream>>>(A, ni, ei, E);
    hipMemsetAsync(A, 0, FEAT * 4, stream);
    node_scatter<<<scatterGrid, 256, 0, stream>>>(E, Binv, ni, ei, A);
    finalize1<<<featGrid, 256, 0, stream>>>(A, Dinv, b1);        // A = h1

    // ---- conv2 ----
    gemm64_wmma<<<gemmGrid, 256, 0, stream>>>(A, W2, B, N_NODES);
    hipMemsetAsync(E, 0, EF * 4, stream);
    edge_scatter<<<scatterGrid, 256, 0, stream>>>(B, ni, ei, E);
    hipMemsetAsync(B, 0, FEAT * 4, stream);
    node_scatter<<<scatterGrid, 256, 0, stream>>>(E, Binv, ni, ei, B);  // B = agg2

    // ---- fuse h = relu(h1 + h2) with pooling, then head GEMM ----
    finalize2_pool<<<featGrid, 256, 0, stream>>>(A, B, Dinv, b2, batch, gsum, gmax);
    head_kernel<<<1, 1024, 0, stream>>>(gsum, gmax, cnt, Wlin, blin, out);
}
